// MultiHeadAttention_65085934404145
// MI455X (gfx1250) — matrix operand order repaired and checked by reading, not compiled, not run on hardware
//
#include <hip/hip_runtime.h>
#include <hip/hip_bf16.h>

typedef __attribute__((ext_vector_type(16))) _Float16 v16h;
typedef __attribute__((ext_vector_type(8)))  float    v8f;

#define HNUM   16
#define DMODEL 1024
#define DKV    64
#define BATCH  4
#define LSEQ   1024
#define MTOT   4096                  // BATCH * LSEQ
#define INV_TEMPER (1.0f / 32.0f)    // 1/sqrt(D_MODEL)
#define NEGINF (-1e9f)

// GEMM kernels: allow full VGPR allocation so the pipelined fragments stay
// in registers; latency is hidden in-wave by the ping-pong buffers.
#define GEMM_BOUNDS __launch_bounds__(256, 1)

// ---------------------------------------------------------------------------
// WMMA fragment loaders from pre-packed f16 — layouts per CDNA5 ISA 7.12.2.
// All loads are contiguous 16B runs per lane -> global_load_b128.
// ---------------------------------------------------------------------------

// A-matrix 16x32 f16 from row-major f16 source (row stride lda)
__device__ __forceinline__ v16h load_a16(const _Float16* __restrict__ A, int lda,
                                         int row0, int k0) {
  const int lane = threadIdx.x & 31;
  const int m    = lane & 15;
  const int koff = (lane < 16) ? 0 : 8;
  const _Float16* p = A + (size_t)(row0 + m) * lda + k0 + koff;
  v16h a;
#pragma unroll
  for (int j = 0; j < 8; ++j) { a[j] = p[j]; a[8 + j] = p[16 + j]; }
  return a;
}

// B-matrix 32x16 f16 from an N-major f16 source (B[k][n] = Bt[n][k]),
// two contiguous 16B runs per lane
__device__ __forceinline__ v16h load_bt16(const _Float16* __restrict__ Bt, int ldb,
                                          int k0, int n0) {
  const int lane = threadIdx.x & 31;
  const int n    = lane & 15;
  const int koff = (lane < 16) ? 0 : 8;
  const _Float16* p = Bt + (size_t)(n0 + n) * ldb + k0 + koff;
  v16h b;
#pragma unroll
  for (int j = 0; j < 8; ++j) { b[j] = p[j]; b[8 + j] = p[16 + j]; }
  return b;
}

__device__ __forceinline__ v8f wmma32(v16h a, v16h b, v8f c) {
  return __builtin_amdgcn_wmma_f32_16x16x32_f16(false, a, false, b,
                                                (short)0, c, false, false);
}

// ---------------------------------------------------------------------------
// Ping-pong pipelined 16xKTOTx64 strip GEMM (2 K-steps per iteration, two
// named buffer sets -> loads land directly in the registers WMMA consumes,
// no rotation moves). Optional pf: global_prefetch_b8 stream.
// ---------------------------------------------------------------------------
template <int KTOT>
__device__ __forceinline__ void gemm_strip(const _Float16* __restrict__ A, int lda,
                                           int row0,
                                           const _Float16* __restrict__ Bt, int ldb,
                                           int n0, v8f acc[4],
                                           const _Float16* pf = nullptr) {
  static_assert((KTOT / 32) % 2 == 0, "even number of 32-wide K-steps required");
  v16h a0 = load_a16(A, lda, row0, 0);
  v16h b0[4];
#pragma unroll
  for (int s3 = 0; s3 < 4; ++s3) b0[s3] = load_bt16(Bt, ldb, 0, n0 + s3 * 16);
  for (int k0 = 0; k0 < KTOT - 64; k0 += 64) {
    // stage k0+32 into buffer 1, compute step k0 from buffer 0
    v16h a1 = load_a16(A, lda, row0, k0 + 32);
    v16h b1[4];
#pragma unroll
    for (int s3 = 0; s3 < 4; ++s3) b1[s3] = load_bt16(Bt, ldb, k0 + 32, n0 + s3 * 16);
    if (pf) __builtin_prefetch(pf + k0 + 256, 0, 1);   // global_prefetch_b8
#pragma unroll
    for (int s3 = 0; s3 < 4; ++s3) acc[s3] = wmma32(a0, b0[s3], acc[s3]);
    // stage k0+64 into buffer 0, compute step k0+32 from buffer 1
    a0 = load_a16(A, lda, row0, k0 + 64);
#pragma unroll
    for (int s3 = 0; s3 < 4; ++s3) b0[s3] = load_bt16(Bt, ldb, k0 + 64, n0 + s3 * 16);
#pragma unroll
    for (int s3 = 0; s3 < 4; ++s3) acc[s3] = wmma32(a1, b1[s3], acc[s3]);
  }
  // epilogue: last two K-steps (buffer 0 holds step KTOT-64)
  {
    v16h a1 = load_a16(A, lda, row0, KTOT - 32);
    v16h b1[4];
#pragma unroll
    for (int s3 = 0; s3 < 4; ++s3) b1[s3] = load_bt16(Bt, ldb, KTOT - 32, n0 + s3 * 16);
#pragma unroll
    for (int s3 = 0; s3 < 4; ++s3) acc[s3] = wmma32(a0, b0[s3], acc[s3]);
#pragma unroll
    for (int s3 = 0; s3 < 4; ++s3) acc[s3] = wmma32(a1, b1[s3], acc[s3]);
  }
}

// ---------------------------------------------------------------------------
// Elementwise f32 -> f16 conversion (same layout)
// ---------------------------------------------------------------------------
__global__ void k_cvt(const float* __restrict__ src, _Float16* __restrict__ dst) {
  const size_t i = (size_t)blockIdx.x * blockDim.x + threadIdx.x;
  dst[i] = (_Float16)src[i];
}

// Head weights (H, DMODEL, DKV) -> fused-transposed f16 WT[n][k],
// n = h*64+kk over 1024, k over DMODEL
__global__ void k_cvt_headw(const float* __restrict__ W, _Float16* __restrict__ WT) {
  const size_t i = (size_t)blockIdx.x * blockDim.x + threadIdx.x; // n*1024 + k
  const int n = (int)(i >> 10), k = (int)(i & 1023);
  WT[i] = (_Float16)W[((size_t)(n >> 6) * DMODEL + k) * DKV + (n & 63)];
}

// ---------------------------------------------------------------------------
// 1) Per-head projections (fused over heads): 4096x1024x1024 GEMM.
//    vmode=0: Out(h,m,kk) f16 (for Q,K). vmode=1: OutT(n,m) f16 (for V).
// ---------------------------------------------------------------------------
__global__ void GEMM_BOUNDS k_head_proj(const _Float16* __restrict__ X,
                                        const _Float16* __restrict__ WT,
                                        _Float16* __restrict__ Out, int vmode) {
  const int wave = threadIdx.x >> 5;
  const int tile = blockIdx.x * 8 + wave;      // 4096 tiles = 256 x 16
  const int tm = tile >> 4, tn = tile & 15;
  const int row0 = tm * 16, n0 = tn * 64;
  v8f acc[4] = {};
  gemm_strip<DMODEL>(X, DMODEL, row0, WT, DMODEL, n0, acc);
  const int lane = threadIdx.x & 31;
  const int mb = row0 + ((lane < 16) ? 0 : 8);
#pragma unroll
  for (int s3 = 0; s3 < 4; ++s3) {
    const int n = n0 + s3 * 16 + (lane & 15);
    if (vmode) {
#pragma unroll
      for (int i = 0; i < 8; ++i)                       // contiguous per lane
        Out[(size_t)n * MTOT + mb + i] = (_Float16)acc[s3][i];
    } else {
      const int h = n >> 6, kk = n & 63;
#pragma unroll
      for (int i = 0; i < 8; ++i)
        Out[((size_t)h * MTOT + mb + i) * DKV + kk] = (_Float16)acc[s3][i];
    }
  }
}

// ---------------------------------------------------------------------------
// 2) Scores: per (h,b)  S = Q(1024x64) @ K^T, scale+mask -> attns (fp32 out)
// ---------------------------------------------------------------------------
__global__ void GEMM_BOUNDS k_scores(const _Float16* __restrict__ qh,
                                     const _Float16* __restrict__ kh,
                                     const unsigned char* __restrict__ mask,
                                     float* __restrict__ attns) {
  const int wave = threadIdx.x >> 5;
  const int tile = blockIdx.x * 8 + wave;      // 64 hb * 1024 tiles
  const int hb = tile >> 10;
  const int t  = tile & 1023;
  const int tq = t >> 4, ts = t & 15;
  const int h = hb >> 2, b = hb & 3;
  const int q0 = tq * 16, s0 = ts * 64;
  const _Float16* Q = qh + ((size_t)h * MTOT + (size_t)b * LSEQ) * DKV;
  const _Float16* K = kh + ((size_t)h * MTOT + (size_t)b * LSEQ) * DKV;
  v8f acc[4] = {};
  gemm_strip<DKV>(Q, DKV, q0, K, DKV, s0, acc);         // B[k][s] = K[s][k]
  const int lane = threadIdx.x & 31;
  const int mb = q0 + ((lane < 16) ? 0 : 8);
  float* Orow = attns + (size_t)hb * LSEQ * LSEQ;
  const unsigned char* Mk = mask + (size_t)b * LSEQ * LSEQ;
#pragma unroll
  for (int s3 = 0; s3 < 4; ++s3) {
    const int s = s0 + s3 * 16 + (lane & 15);
#pragma unroll
    for (int i = 0; i < 8; ++i) {
      const int qq = mb + i;
      float val = acc[s3][i] * INV_TEMPER;
      if (Mk[(size_t)qq * LSEQ + s]) val = NEGINF;
      Orow[(size_t)qq * LSEQ + s] = val;
    }
  }
}

// ---------------------------------------------------------------------------
// 3) Row softmax (fp32, in place in d_out) + f16 shadow copy for attn@V
// ---------------------------------------------------------------------------
__global__ void k_softmax(float* __restrict__ attns, _Float16* __restrict__ a16) {
  float* p = attns + (size_t)blockIdx.x * LSEQ;
  _Float16* p16 = a16 + (size_t)blockIdx.x * LSEQ;
  __shared__ float red[256];
  const int tid = threadIdx.x;
  float vals[4];
  float m = -3.4e38f;
#pragma unroll
  for (int i = 0; i < 4; ++i) { vals[i] = p[tid + i * 256]; m = fmaxf(m, vals[i]); }
  red[tid] = m; __syncthreads();
  for (int s = 128; s > 0; s >>= 1) {
    if (tid < s) red[tid] = fmaxf(red[tid], red[tid + s]);
    __syncthreads();
  }
  m = red[0]; __syncthreads();
  float sum = 0.f;
#pragma unroll
  for (int i = 0; i < 4; ++i) { vals[i] = __expf(vals[i] - m); sum += vals[i]; }
  red[tid] = sum; __syncthreads();
  for (int s = 128; s > 0; s >>= 1) {
    if (tid < s) red[tid] += red[tid + s];
    __syncthreads();
  }
  const float inv = 1.0f / red[0];
#pragma unroll
  for (int i = 0; i < 4; ++i) {
    const float r = vals[i] * inv;
    p[tid + i * 256]   = r;
    p16[tid + i * 256] = (_Float16)r;
  }
}

// ---------------------------------------------------------------------------
// 4) out_h: per (h,b)  O(1024x64) = attns16 @ V; wave does full dv width
// ---------------------------------------------------------------------------
__global__ void GEMM_BOUNDS k_av(const _Float16* __restrict__ a16,
                                 const _Float16* __restrict__ vhT,
                                 _Float16* __restrict__ outc) {
  const int wave = threadIdx.x >> 5;
  const int tile = blockIdx.x * 8 + wave;      // 64 hb * 64 tq
  const int hb = tile >> 6;
  const int tq = tile & 63;
  const int h = hb >> 2, b = hb & 3;
  const int q0 = tq * 16;
  const _Float16* Arow = a16 + (size_t)hb * LSEQ * LSEQ;
  // B[k][kk] = vhT[(h*64+kk)*MTOT + b*LSEQ + k]
  const _Float16* VT = vhT + (size_t)h * DKV * MTOT + (size_t)b * LSEQ;
  v8f acc[4] = {};
  const _Float16* pf = Arow + (size_t)(q0 + (threadIdx.x & 15)) * LSEQ;
  gemm_strip<LSEQ>(Arow, LSEQ, q0, VT, MTOT, 0, acc, pf);
  const int lane = threadIdx.x & 31;
  const int mb = q0 + ((lane < 16) ? 0 : 8);
#pragma unroll
  for (int s3 = 0; s3 < 4; ++s3) {
    const int n = s3 * 16 + (lane & 15);
#pragma unroll
    for (int i = 0; i < 8; ++i)
      outc[((size_t)b * LSEQ + mb + i) * DMODEL + h * DKV + n] = (_Float16)acc[s3][i];
  }
}

// ---------------------------------------------------------------------------
// 5) Output projection: X = outc16 @ proj_w^T + bias + residual (fp32 out)
// ---------------------------------------------------------------------------
__global__ void GEMM_BOUNDS k_proj(const _Float16* __restrict__ outc,
                                   const _Float16* __restrict__ Wp, // Wp[n][k]
                                   const float* __restrict__ bias,
                                   const float* __restrict__ resid,
                                   float* __restrict__ X) {
  const int wave = threadIdx.x >> 5;
  const int tile = blockIdx.x * 8 + wave;      // 256 x 16
  const int tm = tile >> 4, tn = tile & 15;
  const int row0 = tm * 16, n0 = tn * 64;
  v8f acc[4] = {};
  gemm_strip<DMODEL>(outc, DMODEL, row0, Wp, DMODEL, n0, acc);
  const int lane = threadIdx.x & 31;
  const int mb = row0 + ((lane < 16) ? 0 : 8);
#pragma unroll
  for (int s3 = 0; s3 < 4; ++s3) {
    const int n = n0 + s3 * 16 + (lane & 15);
#pragma unroll
    for (int i = 0; i < 8; ++i) {
      const size_t idx = (size_t)(mb + i) * DMODEL + n;
      X[idx] = acc[s3][i] + bias[n] + resid[idx];
    }
  }
}

// ---------------------------------------------------------------------------
// 6) BatchNorm: population stats over (B,L) per channel, then apply
// ---------------------------------------------------------------------------
__global__ void k_bnstats(const float* __restrict__ X,
                          float* __restrict__ mean, float* __restrict__ rstd) {
  const int d = blockIdx.x;
  const int tid = threadIdx.x;
  float s = 0.f, s2 = 0.f;
  for (int m = tid; m < MTOT; m += 256) {
    const float v = X[(size_t)m * DMODEL + d];
    s += v; s2 += v * v;
  }
  __shared__ float rs[256], rs2[256];
  rs[tid] = s; rs2[tid] = s2; __syncthreads();
  for (int t = 128; t > 0; t >>= 1) {
    if (tid < t) { rs[tid] += rs[tid + t]; rs2[tid] += rs2[tid + t]; }
    __syncthreads();
  }
  if (tid == 0) {
    const float mu  = rs[0] / (float)MTOT;
    const float var = rs2[0] / (float)MTOT - mu * mu;
    mean[d] = mu;
    rstd[d] = rsqrtf(var + 1e-5f);
  }
}

__global__ void k_bnapply(const float* __restrict__ X,
                          const float* __restrict__ mean,
                          const float* __restrict__ rstd,
                          const float* __restrict__ gamma,
                          const float* __restrict__ beta,
                          float* __restrict__ out) {
  const size_t idx = (size_t)blockIdx.x * blockDim.x + threadIdx.x;
  const int d = (int)(idx & (DMODEL - 1));
  out[idx] = (X[idx] - mean[d]) * rstd[d] * gamma[d] + beta[d];
}

// ---------------------------------------------------------------------------
extern "C" void kernel_launch(void* const* d_in, const int* in_sizes, int n_in,
                              void* d_out, int out_size, void* d_ws, size_t ws_size,
                              hipStream_t stream) {
  (void)in_sizes; (void)n_in; (void)out_size; (void)ws_size;
  const float* q      = (const float*)d_in[0];
  const float* k      = (const float*)d_in[1];
  const float* v      = (const float*)d_in[2];
  const unsigned char* mask = (const unsigned char*)d_in[3]; // jnp.bool_: 1 B/elem
  const float* w_qs   = (const float*)d_in[4];
  const float* w_ks   = (const float*)d_in[5];
  const float* w_vs   = (const float*)d_in[6];
  const float* proj_w = (const float*)d_in[7];
  const float* proj_b = (const float*)d_in[8];
  const float* gamma  = (const float*)d_in[9];
  const float* beta   = (const float*)d_in[10];

  float* out   = (float*)d_out;
  float* attns = out + (size_t)BATCH * LSEQ * DMODEL;   // second output region

  // --- workspace layout: fp32 first, then f16 region ---
  float* xbuf = (float*)d_ws;                           // 4096x1024
  float* mean = xbuf + (size_t)MTOT * DMODEL;           // 1024
  float* rstd = mean + DMODEL;                          // 1024
  _Float16* hp = (_Float16*)(rstd + DMODEL);
  _Float16* q16   = hp;  hp += (size_t)MTOT * DMODEL;   // 4M halves each
  _Float16* k16   = hp;  hp += (size_t)MTOT * DMODEL;
  _Float16* v16   = hp;  hp += (size_t)MTOT * DMODEL;
  _Float16* wqT   = hp;  hp += (size_t)DMODEL * DMODEL; // 1M halves each
  _Float16* wkT   = hp;  hp += (size_t)DMODEL * DMODEL;
  _Float16* wvT   = hp;  hp += (size_t)DMODEL * DMODEL;
  _Float16* wp16  = hp;  hp += (size_t)DMODEL * DMODEL;
  _Float16* qh16  = hp;  hp += (size_t)HNUM * MTOT * DKV;
  _Float16* kh16  = hp;  hp += (size_t)HNUM * MTOT * DKV;
  _Float16* vhT16 = hp;  hp += (size_t)HNUM * MTOT * DKV;
  _Float16* a16   = hp;  hp += (size_t)HNUM * BATCH * LSEQ * LSEQ; // 64M halves
  _Float16* outc16= hp;

  const dim3 blk(256);  // 8 wave32 waves / block
  // stage operands as f16 (no cvt in GEMM loops, halves operand bytes)
  k_cvt      <<<16384, blk, 0, stream>>>(q, q16);
  k_cvt      <<<16384, blk, 0, stream>>>(k, k16);
  k_cvt      <<<16384, blk, 0, stream>>>(v, v16);
  k_cvt_headw<<<4096,  blk, 0, stream>>>(w_qs, wqT);
  k_cvt_headw<<<4096,  blk, 0, stream>>>(w_ks, wkT);
  k_cvt_headw<<<4096,  blk, 0, stream>>>(w_vs, wvT);
  k_cvt      <<<4096,  blk, 0, stream>>>(proj_w, wp16);
  // WMMA pipeline
  k_head_proj<<<512,   blk, 0, stream>>>(q16, wqT, qh16, 0);
  k_head_proj<<<512,   blk, 0, stream>>>(k16, wkT, kh16, 0);
  k_head_proj<<<512,   blk, 0, stream>>>(v16, wvT, vhT16, 1);
  k_scores   <<<8192,  blk, 0, stream>>>(qh16, kh16, mask, attns);
  k_softmax  <<<65536, blk, 0, stream>>>(attns, a16);
  k_av       <<<512,   blk, 0, stream>>>(a16, vhT16, outc16);
  k_proj     <<<512,   blk, 0, stream>>>(outc16, wp16, proj_b, q, xbuf);
  k_bnstats  <<<1024,  blk, 0, stream>>>(xbuf, mean, rstd);
  k_bnapply  <<<16384, blk, 0, stream>>>(xbuf, mean, rstd, gamma, beta, out);
}
